// StandardGatedRNN_14937896256167
// MI455X (gfx1250) — matrix operand order repaired and checked by reading, not compiled, not run on hardware
//
#include <hip/hip_runtime.h>

// ---------------------------------------------------------------------------
// StandardGatedRNN forward for MI455X (gfx1250, wave32, WMMA bf16)
// B=128, T=512, H=1024, V=1024, I=512.  ~550 GFLOP of GEMM work, weights
// (8.2 MB bf16) resident in the 192 MB L2; per-step A-operands staged in LDS.
//
// GEMM loop order: k0 outer / N-tile inner with persistent accumulators, so
// each LDS A-fragment (8 VGPRs) is loaded once per k0 and fed to 4 WMMAs —
// keeps VGPR pressure ~60 and avoids the scratch spill/reload the compiler
// generated for the tile-outer formulation.
// ---------------------------------------------------------------------------

#define B_  128
#define T_  512
#define H_  1024
#define V_  1024
#define I_  512
#define K1_ 3072          // 3*H gate-input width
#define AP  1032          // padded LDS pitch (bf16 elems) for H-wide tiles
#define HP  520           // padded LDS pitch for I-wide tiles

typedef __bf16          bfv16 __attribute__((ext_vector_type(16)));
typedef float           fv8   __attribute__((ext_vector_type(8)));
typedef unsigned short  usv8  __attribute__((ext_vector_type(8)));

__device__ inline unsigned short f2bf(float f) {
  union { float f; unsigned u; } c; c.f = f;
  unsigned r = (c.u + 0x7FFFu + ((c.u >> 16) & 1u)) >> 16;   // round-nearest-even
  return (unsigned short)r;
}
__device__ inline float bf2f(unsigned short s) {
  union { unsigned u; float f; } c; c.u = ((unsigned)s) << 16;
  return c.f;
}

// A fragment (16-bit, 16x32, wave32): lane&15 = row M; lanes<16 hold K 0-15,
// lanes>=16 hold K 16-31 (matches B fragment).  Two ds_load_b128.
__device__ inline bfv16 load_a_frag(const unsigned short* base, int pitch,
                                    int lane, int k0) {
  const int row = lane & 15;
  const int kb  = k0 + ((lane >> 4) << 4);
  const unsigned short* p = base + row * pitch + kb;
  union { bfv16 v; usv8 h[2]; } u;
  u.h[0] = *reinterpret_cast<const usv8*>(p);
  u.h[1] = *reinterpret_cast<const usv8*>(p + 8);
  return u.v;
}

// B fragment (16-bit, 32x16): lane&15 = col N; lanes 0-15 hold K 0-15,
// lanes 16-31 hold K 16-31.  B column n of (X @ W^T) is row n of W, so this
// is one contiguous 32-byte global load straight out of L2.
__device__ inline bfv16 load_b_frag(const unsigned short* W, int K, int n0,
                                    int lane, int k0) {
  const int n  = n0 + (lane & 15);
  const int kb = k0 + ((lane >> 4) << 4);
  return *reinterpret_cast<const bfv16*>(W + (size_t)n * K + kb);
}

__device__ inline fv8 wmma_bf16(bfv16 a, bfv16 b, fv8 c) {
  // (neg_a, A, neg_b, B, c_mod, C, reuse_a, reuse_b)
  return __builtin_amdgcn_wmma_f32_16x16x32_bf16(false, a, false, b,
                                                 (short)0, c, false, false);
}

// ---------------------------------------------------------------------------
// One recurrence step.  gridDim.x = B/16, blockDim.x = 512 (16 wave32s).
// Workgroup owns 16 batch rows -> M=16 exactly matches the WMMA tile.
// ---------------------------------------------------------------------------
__global__ __launch_bounds__(512) void rnn_step_kernel(
    int t,
    const int*   __restrict__ obs,   const float* __restrict__ vel,
    const float* __restrict__ embed, const float* __restrict__ Wv,
    const float* __restrict__ bv,    const float* __restrict__ b1,
    const float* __restrict__ b2,    const float* __restrict__ b_rec,
    const unsigned short* __restrict__ W1b,  const unsigned short* __restrict__ W2b,
    const unsigned short* __restrict__ Wrecb,const unsigned short* __restrict__ Woutb,
    float* __restrict__ h_state, float* __restrict__ pred)
{
  __shared__ alignas(32) unsigned short sA[16 * AP];     // staged A operand (bf16)
  __shared__ alignas(32) unsigned short sHid[16 * HP];   // relu hidden (bf16)
  __shared__ alignas(32) unsigned short sSig[16 * AP];   // sigma (bf16)
  __shared__ alignas(32) unsigned short sHrec[16 * AP];  // h @ W_rec^T (bf16)

  const int tid  = threadIdx.x;
  const int wave = tid >> 5;
  const int lane = tid & 31;
  const int b0   = blockIdx.x << 4;
  const int n0w  = wave << 6;                 // this wave's 64-col slab (4 tiles)

  // ---- P1: stage h (fp32 -> bf16) into sA --------------------------------
  for (int i = tid; i < 16 * H_; i += 512) {
    int r = i >> 10, c = i & (H_ - 1);
    sA[r * AP + c] = f2bf(h_state[(size_t)(b0 + r) * H_ + c]);
  }
  __syncthreads();

  // ---- P2: hrec = h @ W_rec^T (K=1024); k0 outer, 4 N-tiles inner --------
  {
    fv8 acc[4];
    acc[0] = (fv8)(0.0f); acc[1] = (fv8)(0.0f);
    acc[2] = (fv8)(0.0f); acc[3] = (fv8)(0.0f);
    for (int k0 = 0; k0 < H_; k0 += 32) {
      bfv16 a = load_a_frag(sA, AP, lane, k0);
      #pragma unroll
      for (int tile = 0; tile < 4; ++tile) {
        bfv16 b = load_b_frag(Wrecb, H_, n0w + (tile << 4), lane, k0);
        acc[tile] = wmma_bf16(a, b, acc[tile]);
      }
    }
    #pragma unroll
    for (int tile = 0; tile < 4; ++tile) {
      const int n = n0w + (tile << 4) + (lane & 15);
      for (int i = 0; i < 8; ++i) {
        int row = i + ((lane >> 4) << 3);
        sHrec[row * AP + n] = f2bf(acc[tile][i]);
      }
    }
  }

  // ---- P3..P5: hid = relu([h|oe|ve] @ W1^T + b1), 3 K-chunks of 1024 -----
  // Accumulators persist in VGPRs across LDS restaging; 2 N-tiles per wave.
  fv8 acc1_0 = (fv8)(0.0f), acc1_1 = (fv8)(0.0f);
  for (int chunk = 0; chunk < 3; ++chunk) {
    if (chunk > 0) {
      __syncthreads();                     // all waves done reading prior sA
      if (chunk == 1) {                    // obs embedding gather
        for (int i = tid; i < 16 * H_; i += 512) {
          int r = i >> 10, c = i & (H_ - 1);
          int idx = obs[(size_t)(b0 + r) * T_ + t];
          sA[r * AP + c] = f2bf(embed[(size_t)idx * H_ + c]);
        }
      } else {                             // velocity embedding
        for (int i = tid; i < 16 * H_; i += 512) {
          int r = i >> 10, c = i & (H_ - 1);
          float vv = vel[(size_t)(b0 + r) * T_ + t];
          sA[r * AP + c] = f2bf(vv * Wv[c] + bv[c]);
        }
      }
      __syncthreads();
    }
    const unsigned short* Wc = W1b + (size_t)chunk * H_;  // column offset in K1
    for (int k0 = 0; k0 < H_; k0 += 32) {
      bfv16 a  = load_a_frag(sA, AP, lane, k0);
      bfv16 g0 = load_b_frag(Wc, K1_, ((wave << 1) + 0) << 4, lane, k0);
      acc1_0 = wmma_bf16(a, g0, acc1_0);
      bfv16 g1 = load_b_frag(Wc, K1_, ((wave << 1) + 1) << 4, lane, k0);
      acc1_1 = wmma_bf16(a, g1, acc1_1);
    }
  }
  // bias + relu -> sHid
  #pragma unroll
  for (int tile = 0; tile < 2; ++tile) {
    const fv8 acc = tile ? acc1_1 : acc1_0;
    const int n = (((wave << 1) + tile) << 4) + (lane & 15);
    for (int i = 0; i < 8; ++i) {
      int row = i + ((lane >> 4) << 3);
      float v = acc[i] + b1[n];
      sHid[row * HP + n] = f2bf(v > 0.f ? v : 0.f);
    }
  }
  __syncthreads();

  // ---- P6: sigma = sigmoid(hid @ W2^T + b2), K=512; k0 outer -------------
  {
    fv8 acc[4];
    acc[0] = (fv8)(0.0f); acc[1] = (fv8)(0.0f);
    acc[2] = (fv8)(0.0f); acc[3] = (fv8)(0.0f);
    for (int k0 = 0; k0 < I_; k0 += 32) {
      bfv16 a = load_a_frag(sHid, HP, lane, k0);
      #pragma unroll
      for (int tile = 0; tile < 4; ++tile) {
        bfv16 b = load_b_frag(W2b, I_, n0w + (tile << 4), lane, k0);
        acc[tile] = wmma_bf16(a, b, acc[tile]);
      }
    }
    #pragma unroll
    for (int tile = 0; tile < 4; ++tile) {
      const int n = n0w + (tile << 4) + (lane & 15);
      for (int i = 0; i < 8; ++i) {
        int row = i + ((lane >> 4) << 3);
        float s = 1.f / (1.f + __expf(-(acc[tile][i] + b2[n])));
        sSig[row * AP + n] = f2bf(s);
      }
    }
  }
  __syncthreads();

  // ---- P7: z = sig*hrec + (1-sig)*(oe+ve) + b_rec; LN; leaky-relu --------
  // wave w owns batch row w; x recomputed in fp32 from L2-resident embed.
  {
    const int row = wave;
    const int b   = b0 + row;
    const int idx = obs[(size_t)b * T_ + t];
    const float* er = embed + (size_t)idx * H_;
    const float  vv = vel[(size_t)b * T_ + t];
    float z[32];
    float s1 = 0.f, s2 = 0.f;
    for (int j = 0; j < 32; ++j) {
      int c = lane + (j << 5);
      float x  = er[c] + vv * Wv[c] + bv[c];
      float sg = bf2f(sSig[row * AP + c]);
      float hr = bf2f(sHrec[row * AP + c]);
      float zz = sg * hr + (1.f - sg) * x + b_rec[c];
      z[j] = zz; s1 += zz; s2 += zz * zz;
    }
    for (int m = 16; m > 0; m >>= 1) {
      s1 += __shfl_xor(s1, m, 32);
      s2 += __shfl_xor(s2, m, 32);
    }
    const float mean = s1 * (1.f / H_);
    const float var  = s2 * (1.f / H_) - mean * mean;
    const float rstd = rsqrtf(var + 1e-5f);
    for (int j = 0; j < 32; ++j) {
      int c = lane + (j << 5);
      float zn = (z[j] - mean) * rstd;
      float hn = zn >= 0.f ? zn : 0.01f * zn;
      h_state[(size_t)b * H_ + c] = hn;     // fp32 carry for next step
      sA[row * AP + c] = f2bf(hn);          // bf16 A operand for GEMMout
    }
  }
  __syncthreads();

  // ---- P8: pred = h_new @ W_out^T (K=1024); k0 outer, 4 N-tiles inner ----
  {
    fv8 acc[4];
    acc[0] = (fv8)(0.0f); acc[1] = (fv8)(0.0f);
    acc[2] = (fv8)(0.0f); acc[3] = (fv8)(0.0f);
    for (int k0 = 0; k0 < H_; k0 += 32) {
      if (k0 + 32 < H_)   // gfx1250 global_prefetch_b8 of next W_out slab
        __builtin_prefetch(Woutb + (size_t)n0w * H_ + k0 + 32, 0, 3);
      bfv16 a = load_a_frag(sA, AP, lane, k0);
      #pragma unroll
      for (int tile = 0; tile < 4; ++tile) {
        bfv16 b = load_b_frag(Woutb, H_, n0w + (tile << 4), lane, k0);
        acc[tile] = wmma_bf16(a, b, acc[tile]);
      }
    }
    #pragma unroll
    for (int tile = 0; tile < 4; ++tile) {
      const int n = n0w + (tile << 4) + (lane & 15);
      for (int i = 0; i < 8; ++i) {
        int row = i + ((lane >> 4) << 3);
        pred[((size_t)(b0 + row) * T_ + t) * V_ + n] = acc[tile][i];
      }
    }
  }
}

// ---------------------------------------------------------------------------
// Prep / epilogue kernels
// ---------------------------------------------------------------------------
__global__ __launch_bounds__(256) void cvt_bf16_kernel(
    const float* __restrict__ s, unsigned short* __restrict__ d, int n) {
  int i = blockIdx.x * 256 + threadIdx.x;
  if (i < n) d[i] = f2bf(s[i]);
}

__global__ __launch_bounds__(256) void init_kernel(
    const float* __restrict__ h_init, const int* __restrict__ lens,
    float* __restrict__ h_state, float* __restrict__ mask_out,
    float* __restrict__ loss_sum) {
  int i = blockIdx.x * 256 + threadIdx.x;
  if (i == 0) loss_sum[0] = 0.f;
  if (i < B_ * H_) h_state[i] = h_init[i & (H_ - 1)];
  if (i < B_ * T_) {
    int b = i >> 9, t = i & (T_ - 1);
    mask_out[i] = (t >= lens[b] - 1) ? 1.f : 0.f;
  }
}

__global__ __launch_bounds__(256) void loss_accum_kernel(
    const float* __restrict__ pred, const int* __restrict__ target,
    const int* __restrict__ lens, float* __restrict__ loss_sum) {
  const int bt = blockIdx.x;
  const int b = bt >> 9, t = bt & (T_ - 1);
  if (t < lens[b] - 1) return;                 // masked token
  const float* row = pred + (size_t)bt * V_;
  __shared__ float sred[8];
  const int wave = threadIdx.x >> 5, lane = threadIdx.x & 31;

  float m = -3.0e38f;
  for (int c = threadIdx.x; c < V_; c += 256) m = fmaxf(m, row[c]);
  for (int k = 16; k > 0; k >>= 1) m = fmaxf(m, __shfl_xor(m, k, 32));
  if (lane == 0) sred[wave] = m;
  __syncthreads();
  if (threadIdx.x == 0) {
    float mm = sred[0];
    for (int w = 1; w < 8; ++w) mm = fmaxf(mm, sred[w]);
    sred[0] = mm;
  }
  __syncthreads();
  m = sred[0];
  __syncthreads();

  float s = 0.f;
  for (int c = threadIdx.x; c < V_; c += 256) s += __expf(row[c] - m);
  for (int k = 16; k > 0; k >>= 1) s += __shfl_xor(s, k, 32);
  if (lane == 0) sred[wave] = s;
  __syncthreads();
  if (threadIdx.x == 0) {
    float ss = 0.f;
    for (int w = 0; w < 8; ++w) ss += sred[w];
    float per = __logf(ss) + m - row[target[bt]];
    atomicAdd(loss_sum, per);
  }
}

__global__ void finalize_kernel(const float* __restrict__ loss_sum,
                                const int* __restrict__ lens,
                                float* __restrict__ out_loss) {
  if (threadIdx.x == 0 && blockIdx.x == 0) {
    float cnt = 0.f;
    for (int b = 0; b < B_; ++b) cnt += (float)(T_ - lens[b] + 1);
    out_loss[0] = loss_sum[0] / (cnt + 1e-8f);
  }
}

// ---------------------------------------------------------------------------
extern "C" void kernel_launch(void* const* d_in, const int* in_sizes, int n_in,
                              void* d_out, int out_size, void* d_ws, size_t ws_size,
                              hipStream_t stream) {
  (void)in_sizes; (void)n_in; (void)out_size; (void)ws_size;
  const int*   obs    = (const int*)  d_in[0];
  const float* vel    = (const float*)d_in[1];
  const int*   tgt    = (const int*)  d_in[2];
  const int*   lens   = (const int*)  d_in[3];
  const float* embed  = (const float*)d_in[4];
  const float* Wv     = (const float*)d_in[5];
  const float* bv     = (const float*)d_in[6];
  const float* W1     = (const float*)d_in[7];
  const float* b1     = (const float*)d_in[8];
  const float* W2     = (const float*)d_in[9];
  const float* b2     = (const float*)d_in[10];
  const float* Wrec   = (const float*)d_in[11];
  const float* Wout   = (const float*)d_in[12];
  const float* h_init = (const float*)d_in[13];
  const float* b_rec  = (const float*)d_in[14];

  // d_out = [loss(1) | predictions(B*T*V) | mask(B*T)]  (fp32)
  float* out_loss = (float*)d_out;
  float* pred     = out_loss + 1;
  float* mask_out = pred + (size_t)B_ * T_ * V_;

  // Workspace: bf16 weights + h_state + loss accumulator (~8.8 MB)
  char* ws = (char*)d_ws;
  unsigned short* W1b   = (unsigned short*)ws; ws += (size_t)I_ * K1_ * 2;
  unsigned short* W2b   = (unsigned short*)ws; ws += (size_t)H_ * I_  * 2;
  unsigned short* Wrecb = (unsigned short*)ws; ws += (size_t)H_ * H_  * 2;
  unsigned short* Woutb = (unsigned short*)ws; ws += (size_t)V_ * H_  * 2;
  float* h_state        = (float*)ws;          ws += (size_t)B_ * H_  * 4;
  float* loss_sum       = (float*)ws;

  init_kernel<<<(B_ * H_ + 255) / 256, 256, 0, stream>>>(h_init, lens, h_state,
                                                         mask_out, loss_sum);
  cvt_bf16_kernel<<<(I_ * K1_ + 255) / 256, 256, 0, stream>>>(W1,   W1b,   I_ * K1_);
  cvt_bf16_kernel<<<(H_ * I_  + 255) / 256, 256, 0, stream>>>(W2,   W2b,   H_ * I_);
  cvt_bf16_kernel<<<(H_ * H_  + 255) / 256, 256, 0, stream>>>(Wrec, Wrecb, H_ * H_);
  cvt_bf16_kernel<<<(V_ * H_  + 255) / 256, 256, 0, stream>>>(Wout, Woutb, V_ * H_);

  for (int t = 0; t < T_; ++t) {
    rnn_step_kernel<<<B_ / 16, 512, 0, stream>>>(
        t, obs, vel, embed, Wv, bv, b1, b2, b_rec,
        W1b, W2b, Wrecb, Woutb, h_state, pred);
  }

  loss_accum_kernel<<<B_ * T_, 256, 0, stream>>>(pred, tgt, lens, loss_sum);
  finalize_kernel<<<1, 1, 0, stream>>>(loss_sum, lens, out_loss);
}
